// ResLSTM_1975684956896
// MI455X (gfx1250) — matrix operand order repaired and checked by reading, not compiled, not run on hardware
//
#include <hip/hip_runtime.h>

#define TT 512
#define BB 128
#define INP 512
#define HH 1024
#define EPSF 1e-5f

typedef __attribute__((ext_vector_type(16))) __bf16 v16bf;
typedef __attribute__((ext_vector_type(8)))  float  v8f;
typedef __attribute__((ext_vector_type(4)))  unsigned v4u;
typedef __attribute__((ext_vector_type(8)))  int    v8i_t;
typedef __attribute__((ext_vector_type(4)))  int    v4i_t;

union BFrag { v16bf v; unsigned u[8]; };
union BF2   { __bf16 h[2]; unsigned u; };

__device__ __forceinline__ unsigned short f2bf(float f) {
  BF2 t; t.h[0] = (__bf16)f; t.h[1] = (__bf16)0.0f;
  return (unsigned short)(t.u & 0xffffu);
}
__device__ __forceinline__ unsigned pack2(float a, float b) {
  BF2 t; t.h[0] = (__bf16)a; t.h[1] = (__bf16)b;
  return t.u;
}
__device__ __forceinline__ float sigm(float x) { return 1.0f / (1.0f + __expf(-x)); }

// ---------------------------------------------------------------------------
// Tensor Data Mover: DMA a [tile_x elems] x [tile_y rows] x [tile_z planes]
// bf16 tile from global into LDS (contiguous, row-major tile order).
// D# layout per CDNA5 ISA ch.8: group0 = count/lds/global/type,
// group1 = data_size + dims/strides, group2 = dim2. Issued by one wave
// (TDM ignores EXEC; a single issue serves the whole workgroup).
// Toolchain here is the 6-arg builtin variant (clang-23 / therock-10.0).
// ---------------------------------------------------------------------------
__device__ __forceinline__ void tdm_load_tile(unsigned lds_addr, const void* gptr,
                                              int tile_x, int tile_y, int tile_z,
                                              unsigned stride_y_elems,
                                              unsigned long long stride_z_elems)
{
  unsigned long long ga = (unsigned long long)(size_t)gptr;
  v4u g0 = {0u, 0u, 0u, 0u};
  g0[0] = 1u;                                   // count = 1 valid descriptor
  g0[1] = lds_addr;                             // LDS byte address
  g0[2] = (unsigned)ga;                         // global_addr[31:0]
  g0[3] = (unsigned)((ga >> 32) & 0x1ffffffull) // global_addr[56:32]
          | (2u << 30);                         // type = 2 ("image")
  v8i_t g1 = {0, 0, 0, 0, 0, 0, 0, 0};
  g1[0] = (int)(1u << 16);                      // data_size = 1 -> 2 bytes
  g1[1] = (int)((unsigned)tile_x << 16);        // tensor_dim0[15:0]
  g1[2] = (int)((unsigned)tile_y << 16);        // tensor_dim1[15:0]
  g1[3] = (int)((unsigned)tile_x << 16);        // tile_dim0
  g1[4] = (int)((unsigned)tile_y                // tile_dim1
          | ((unsigned)tile_z << 16));          // tile_dim2 (0 = 2D)
  g1[5] = (int)stride_y_elems;                  // tensor_dim0_stride[31:0]
  g1[6] = (int)((unsigned)(stride_z_elems & 0xffffull) << 16); // dim1_stride[15:0]
  g1[7] = (int)(unsigned)((stride_z_elems >> 16) & 0xffffffffull); // dim1_stride[47:16]
  v4i_t g2 = {0, 0, 0, 0};
  g2[0] = tile_z;                               // tensor_dim2
  v4i_t g3 = {0, 0, 0, 0};
  v8i_t g4 = {0, 0, 0, 0, 0, 0, 0, 0};          // extra group (6-arg toolchain form)
  __builtin_amdgcn_tensor_load_to_lds(g0, g1, g2, g3, g4, 0);
}

// ---------------------------------------------------------------------------
// Fused gates GEMM + LSTM cell.
// gates[B,4H] = [X ; H_prev] @ Wcat^T + bias   (bf16 WMMA, f32 accum)
// Block: 256 threads (8 waves). blockIdx.x = j-tile (16 hidden columns).
// B tiles arrive via TDM; A tiles are converted f32->bf16 into LDS by VALU.
// ---------------------------------------------------------------------------
template<int KX>
__global__ __launch_bounds__(256)
void gates_cell_kernel(const float* __restrict__ X,    // [B,KX]
                       const float* __restrict__ Hst,  // [B,H]
                       const unsigned* __restrict__ W2,// bf16 Wcat as u32 pairs, row stride KX+H
                       const float* __restrict__ bias, // [4H]
                       float* __restrict__ cSt,
                       float* __restrict__ tanhcB,
                       float* __restrict__ ogB,
                       float* __restrict__ psum,
                       float* __restrict__ psq)
{
  constexpr int KT = KX + HH;
  constexpr int NC = KT / 64;
  __shared__ unsigned sA[128 * 32];   // [row][kpair] 16KB
  __shared__ unsigned sB[4 * 16 * 32];// [gate][col][kpair] 8KB (TDM destination)

  const int tid  = threadIdx.x;
  const int lane = tid & 31;
  const int wave = tid >> 5;
  const int ml   = lane & 15;
  const int half = lane >> 4;
  const int mbase = wave * 16;
  const int jbase = blockIdx.x * 16;
  const unsigned sB_lds = (unsigned)(size_t)(&sB[0]);

  v8f acc[4];
#pragma unroll
  for (int g = 0; g < 4; ++g)
#pragma unroll
    for (int i = 0; i < 8; ++i) acc[g][i] = 0.0f;

  for (int kc = 0; kc < NC; ++kc) {
    __syncthreads();   // previous sA/sB fully consumed
    if (wave == 0) {
      // 3D tile: 64 K-elems x 16 rows (stride KT) x 4 gates (stride H*KT)
      const unsigned short* gsrc =
          (const unsigned short*)W2 + ((size_t)jbase * KT + (size_t)kc * 64);
      tdm_load_tile(sB_lds, gsrc, 64, 16, 4, (unsigned)KT,
                    (unsigned long long)HH * (unsigned long long)KT);
    }
    // stage A chunk [128 x 64] with inline f32->bf16 conversion
    for (int i = tid; i < 128 * 32; i += 256) {
      int row = i >> 5, p = i & 31;
      int k = kc * 64 + p * 2;
      float f0, f1;
      if (k < KX) { f0 = X[row * KX + k]; f1 = X[row * KX + k + 1]; }
      else        { int kk = k - KX; f0 = Hst[row * HH + kk]; f1 = Hst[row * HH + kk + 1]; }
      sA[i] = pack2(f0, f1);
    }
    if (wave == 0) __builtin_amdgcn_s_wait_tensorcnt(0);
    __syncthreads();
#pragma unroll
    for (int ks = 0; ks < 2; ++ks) {
      BFrag a;
#pragma unroll
      for (int v = 0; v < 8; ++v) {
        int p = ks * 16 + half * 8 + v;
        a.u[v] = sA[(mbase + ml) * 32 + p];
      }
#pragma unroll
      for (int g = 0; g < 4; ++g) {
        BFrag b;
#pragma unroll
        for (int v = 0; v < 8; ++v) {
          int p = ks * 16 + half * 8 + v;
          b.u[v] = sB[(g * 16 + ml) * 32 + p];
        }
        acc[g] = __builtin_amdgcn_wmma_f32_16x16x32_bf16(
            false, a.v, false, b.v, (short)0, acc[g], false, false);
      }
    }
  }

  // LSTM cell epilogue (C layout: col n = lane%16, row m = v + 8*half)
  const int j = jbase + ml;
  const float bI = bias[0 * HH + j], bF = bias[1 * HH + j];
  const float bC = bias[2 * HH + j], bO = bias[3 * HH + j];
#pragma unroll
  for (int v = 0; v < 8; ++v) {
    int m = mbase + v + 8 * half;
    float ig = acc[0][v] + bI;
    float fg = acc[1][v] + bF;
    float cg = acc[2][v] + bC;
    float og = acc[3][v] + bO;
    size_t idx = (size_t)m * HH + j;
    float cnew = sigm(fg) * cSt[idx] + sigm(ig) * tanhf(cg);
    cSt[idx] = cnew;
    float th = tanhf(cnew);
    tanhcB[idx] = th;
    ogB[idx] = sigm(og);
    float s = th, q = th * th;
#pragma unroll
    for (int d = 1; d < 16; d <<= 1) {
      s += __shfl_xor(s, d, 32);
      q += __shfl_xor(q, d, 32);
    }
    if (ml == 0) { psum[blockIdx.x * 128 + m] = s; psq[blockIdx.x * 128 + m] = q; }
  }
}

// ---------------------------------------------------------------------------
// Layer-0 projected residual: resraw[B,H] = x_t @ W_ir^T  (+LN partial sums)
// ---------------------------------------------------------------------------
__global__ __launch_bounds__(256)
void res_gemm_kernel(const float* __restrict__ X,     // [B,512]
                     const unsigned* __restrict__ W2, // bf16 Wir as u32 pairs
                     float* __restrict__ resraw,
                     float* __restrict__ psum,
                     float* __restrict__ psq)
{
  constexpr int KT = INP;
  constexpr int NC = KT / 64;
  __shared__ unsigned sA[128 * 32];
  __shared__ unsigned sB[16 * 32];

  const int tid  = threadIdx.x;
  const int lane = tid & 31;
  const int wave = tid >> 5;
  const int ml   = lane & 15;
  const int half = lane >> 4;
  const int mbase = wave * 16;
  const int jbase = blockIdx.x * 16;
  const unsigned sB_lds = (unsigned)(size_t)(&sB[0]);

  v8f acc;
#pragma unroll
  for (int i = 0; i < 8; ++i) acc[i] = 0.0f;

  for (int kc = 0; kc < NC; ++kc) {
    __syncthreads();
    if (wave == 0) {
      // 2D tile: 64 K-elems x 16 rows (stride KT)
      const unsigned short* gsrc =
          (const unsigned short*)W2 + ((size_t)jbase * KT + (size_t)kc * 64);
      tdm_load_tile(sB_lds, gsrc, 64, 16, 0, (unsigned)KT, 0ull);
    }
    for (int i = tid; i < 128 * 32; i += 256) {
      int row = i >> 5, p = i & 31;
      int k = kc * 64 + p * 2;
      sA[i] = pack2(X[row * KT + k], X[row * KT + k + 1]);
    }
    if (wave == 0) __builtin_amdgcn_s_wait_tensorcnt(0);
    __syncthreads();
#pragma unroll
    for (int ks = 0; ks < 2; ++ks) {
      BFrag a, b;
#pragma unroll
      for (int v = 0; v < 8; ++v) {
        int pa = ks * 16 + half * 8 + v;
        a.u[v] = sA[(mbase + ml) * 32 + pa];
        int pb = ks * 16 + half * 8 + v;
        b.u[v] = sB[ml * 32 + pb];
      }
      acc = __builtin_amdgcn_wmma_f32_16x16x32_bf16(
          false, a.v, false, b.v, (short)0, acc, false, false);
    }
  }

  const int j = jbase + ml;
#pragma unroll
  for (int v = 0; v < 8; ++v) {
    int m = mbase + v + 8 * half;
    float val = acc[v];
    resraw[(size_t)m * HH + j] = val;
    float s = val, q = val * val;
#pragma unroll
    for (int d = 1; d < 16; d <<= 1) {
      s += __shfl_xor(s, d, 32);
      q += __shfl_xor(q, d, 32);
    }
    if (ml == 0) { psum[blockIdx.x * 128 + m] = s; psq[blockIdx.x * 128 + m] = q; }
  }
}

// ---------------------------------------------------------------------------
// Finalize: LN reductions + h = sigmoid(og) * (LN(tanh c) + res)
// ---------------------------------------------------------------------------
template<int LAYER>
__global__ __launch_bounds__(256)
void finalize_kernel(const float* __restrict__ tanhcB, const float* __restrict__ ogB,
                     const float* __restrict__ psumT, const float* __restrict__ psqT,
                     const float* __restrict__ psumR, const float* __restrict__ psqR,
                     const float* __restrict__ hsumIn, const float* __restrict__ hsqIn,
                     const float* __restrict__ resSrc,
                     const float* __restrict__ g_ry, const float* __restrict__ b_ry,
                     const float* __restrict__ g_res, const float* __restrict__ b_res,
                     float* __restrict__ hOut, float* __restrict__ yOut,
                     float* __restrict__ hsumOut, float* __restrict__ hsqOut)
{
  const int b = blockIdx.x;
  const int tid = threadIdx.x;
  __shared__ float red[4][64];
  __shared__ float stats[4];
  if (tid < 64) {
    red[0][tid] = psumT[tid * 128 + b];
    red[1][tid] = psqT[tid * 128 + b];
    if (LAYER == 0) {
      red[2][tid] = psumR[tid * 128 + b];
      red[3][tid] = psqR[tid * 128 + b];
    }
  }
  __syncthreads();
  if (tid == 0) {
    float s0 = 0, s1 = 0, s2 = 0, s3 = 0;
    for (int i = 0; i < 64; ++i) {
      s0 += red[0][i]; s1 += red[1][i];
      if (LAYER == 0) { s2 += red[2][i]; s3 += red[3][i]; }
    }
    float mT = s0 / HH;
    float vT = s1 / HH - mT * mT;
    stats[0] = mT; stats[1] = rsqrtf(vT + EPSF);
    float rs, rq;
    if (LAYER == 0) { rs = s2; rq = s3; }
    else            { rs = hsumIn[b]; rq = hsqIn[b]; }
    float mR = rs / HH;
    float vR = rq / HH - mR * mR;
    stats[2] = mR; stats[3] = rsqrtf(vR + EPSF);
  }
  __syncthreads();
  const float mT = stats[0], rT = stats[1], mR = stats[2], rR = stats[3];
  float hs = 0.0f, hq = 0.0f;
#pragma unroll
  for (int it = 0; it < 4; ++it) {
    int j = tid + it * 256;
    size_t idx = (size_t)b * HH + j;
    float th = tanhcB[idx];
    float ry = (th - mT) * rT * g_ry[j] + b_ry[j];
    float rv = resSrc[idx];
    float rn = (rv - mR) * rR * g_res[j] + b_res[j];
    float h = ogB[idx] * (ry + rn);
    hOut[idx] = h;
    if (LAYER == 1) yOut[(size_t)b * HH + j] = h;
    hs += h; hq += h * h;
  }
  if (LAYER == 0) {
    __shared__ float r1[256], r2[256];
    r1[tid] = hs; r2[tid] = hq;
    __syncthreads();
    for (int s = 128; s > 0; s >>= 1) {
      if (tid < s) { r1[tid] += r1[tid + s]; r2[tid] += r2[tid + s]; }
      __syncthreads();
    }
    if (tid == 0) { hsumOut[b] = r1[0]; hsqOut[b] = r2[0]; }
  }
}

// ---------------------------------------------------------------------------
// Utility kernels
// ---------------------------------------------------------------------------
__global__ void pack_weights_kernel(const float* __restrict__ Wa, const float* __restrict__ Wb,
                                    int KA, int KB, unsigned short* __restrict__ dst,
                                    long long total)
{
  const int KT = KA + KB;
  for (long long i = (long long)blockIdx.x * blockDim.x + threadIdx.x; i < total;
       i += (long long)gridDim.x * blockDim.x) {
    long long row = i / KT;
    int k = (int)(i % KT);
    float f = (k < KA) ? Wa[row * KA + k] : Wb[row * KB + (k - KA)];
    dst[i] = f2bf(f);
  }
}

__global__ void zero_kernel(float* __restrict__ p, long long n)
{
  for (long long i = (long long)blockIdx.x * blockDim.x + threadIdx.x; i < n;
       i += (long long)gridDim.x * blockDim.x) p[i] = 0.0f;
}

__global__ void copy_states_kernel(const float* __restrict__ h0, const float* __restrict__ h1,
                                   const float* __restrict__ c0, const float* __restrict__ c1,
                                   float* __restrict__ out)
{
  const long long BH = (long long)BB * HH;
  for (long long i = (long long)blockIdx.x * blockDim.x + threadIdx.x; i < BH;
       i += (long long)gridDim.x * blockDim.x) {
    out[i]          = h0[i];
    out[BH + i]     = h1[i];
    out[2 * BH + i] = c0[i];
    out[3 * BH + i] = c1[i];
  }
}

// ---------------------------------------------------------------------------
extern "C" void kernel_launch(void* const* d_in, const int* in_sizes, int n_in,
                              void* d_out, int out_size, void* d_ws, size_t ws_size,
                              hipStream_t stream)
{
  (void)in_sizes; (void)n_in; (void)out_size; (void)ws_size;
  const float* x     = (const float*)d_in[0];
  const float* Wih0  = (const float*)d_in[1];
  const float* Whh0  = (const float*)d_in[2];
  const float* bias0 = (const float*)d_in[3];
  const float* Wir0  = (const float*)d_in[4];
  const float* gry0  = (const float*)d_in[5];
  const float* bry0  = (const float*)d_in[6];
  const float* gres0 = (const float*)d_in[7];
  const float* bres0 = (const float*)d_in[8];
  const float* Wih1  = (const float*)d_in[9];
  const float* Whh1  = (const float*)d_in[10];
  const float* bias1 = (const float*)d_in[11];
  const float* gry1  = (const float*)d_in[12];
  const float* bry1  = (const float*)d_in[13];
  const float* gres1 = (const float*)d_in[14];
  const float* bres1 = (const float*)d_in[15];
  float* out = (float*)d_out;

  char* w = (char*)d_ws;
  size_t off = 0;
  auto alloc = [&](size_t bytes) -> void* {
    void* p = w + off;
    off = (off + bytes + 255) & ~(size_t)255;
    return p;
  };
  const size_t BH = (size_t)BB * HH;
  unsigned short* WC0 = (unsigned short*)alloc((size_t)4 * HH * (INP + HH) * 2);
  unsigned short* WC1 = (unsigned short*)alloc((size_t)4 * HH * (HH + HH) * 2);
  unsigned short* WIR = (unsigned short*)alloc((size_t)HH * INP * 2);
  float* h0     = (float*)alloc(BH * 4);   // h0,c0,h1,c1 contiguous for one zero pass
  float* c0     = (float*)alloc(BH * 4);
  float* h1     = (float*)alloc(BH * 4);
  float* c1     = (float*)alloc(BH * 4);
  float* tanhc0 = (float*)alloc(BH * 4);
  float* og0    = (float*)alloc(BH * 4);
  float* tanhc1 = (float*)alloc(BH * 4);
  float* og1    = (float*)alloc(BH * 4);
  float* resraw = (float*)alloc(BH * 4);
  float* psT0   = (float*)alloc(64 * 128 * 4);
  float* psqT0  = (float*)alloc(64 * 128 * 4);
  float* psR    = (float*)alloc(64 * 128 * 4);
  float* psqR   = (float*)alloc(64 * 128 * 4);
  float* psT1   = (float*)alloc(64 * 128 * 4);
  float* psqT1  = (float*)alloc(64 * 128 * 4);
  float* hsum0  = (float*)alloc(128 * 4);
  float* hsq0   = (float*)alloc(128 * 4);

  pack_weights_kernel<<<4096, 256, 0, stream>>>(Wih0, Whh0, INP, HH, WC0,
                                                (long long)4 * HH * (INP + HH));
  pack_weights_kernel<<<4096, 256, 0, stream>>>(Wih1, Whh1, HH, HH, WC1,
                                                (long long)4 * HH * (HH + HH));
  pack_weights_kernel<<<1024, 256, 0, stream>>>(Wir0, Wir0, INP, 0, WIR,
                                                (long long)HH * INP);
  zero_kernel<<<512, 256, 0, stream>>>(h0, (long long)4 * BH); // h0,c0,h1,c1

  for (int t = 0; t < TT; ++t) {
    const float* Xt = x + (size_t)t * BB * INP;
    gates_cell_kernel<INP><<<64, 256, 0, stream>>>(
        Xt, h0, (const unsigned*)WC0, bias0, c0, tanhc0, og0, psT0, psqT0);
    res_gemm_kernel<<<64, 256, 0, stream>>>(Xt, (const unsigned*)WIR, resraw, psR, psqR);
    finalize_kernel<0><<<128, 256, 0, stream>>>(
        tanhc0, og0, psT0, psqT0, psR, psqR, nullptr, nullptr, resraw,
        gry0, bry0, gres0, bres0, h0, nullptr, hsum0, hsq0);
    gates_cell_kernel<HH><<<64, 256, 0, stream>>>(
        h0, h1, (const unsigned*)WC1, bias1, c1, tanhc1, og1, psT1, psqT1);
    finalize_kernel<1><<<128, 256, 0, stream>>>(
        tanhc1, og1, psT1, psqT1, nullptr, nullptr, hsum0, hsq0, h0,
        gry1, bry1, gres1, bres1, h1, out + (size_t)t * BH, nullptr, nullptr);
  }
  copy_states_kernel<<<256, 256, 0, stream>>>(h0, h1, c0, c1, out + (size_t)TT * BH);
}
